// ga_refinement_53549652246648
// MI455X (gfx1250) — matrix operand order repaired and checked by reading, not compiled, not run on hardware
//
#include <hip/hip_runtime.h>
#include <hip/hip_bf16.h>

// ---------------------------------------------------------------------------
// GA-attention + GUnpool + GBottleneck mesh refinement for MI455X (gfx1250).
// Strategy:
//   * All big GEMMs (h @ W) via v_wmma_f32_16x16x32_bf16 (wave32 WMMA, fp32 acc)
//   * adj is 6-nnz/row row-normalized: extracted once to fixed-width CSR and
//     applied as an exact fp32 gather (avoids 520 GFLOP of dense adj GEMMs)
//   * Epilogues (bias/ReLU/0.5*residual/softmax/coord head) in fp32
// ---------------------------------------------------------------------------

typedef __attribute__((ext_vector_type(16))) __bf16 v16bf;
typedef __attribute__((ext_vector_type(8)))  __bf16 v8bf;
typedef __attribute__((ext_vector_type(8)))  float  v8f;

namespace {
constexpr int B_    = 16;
constexpr int N2_   = 618;
constexpr int N3_   = 2466;
constexpr int FEAT_ = 963;
constexpr int HID_  = 192;
constexpr int EMB_  = 8;
constexpr int IN_DIM_ = FEAT_ + HID_ + EMB_;   // 1163
constexpr int KPAD_   = 1184;                  // 1163 rounded up to K-step 32
constexpr int NBLK_   = 6;
constexpr int MAXD_   = 16;                    // max nnz kept per adj row (actual <= 6)
}

// ---------------------------------------------------------------------------
// 1) q/k/v projection: mv = [0, x2.x, x2.y, x2.z, 0,0,0,0] @ W + b
// ---------------------------------------------------------------------------
__global__ void k_qkv(const float* __restrict__ x2,
                      const float* __restrict__ Wq, const float* __restrict__ bq,
                      const float* __restrict__ Wk, const float* __restrict__ bk,
                      const float* __restrict__ Wv, const float* __restrict__ bv,
                      float* __restrict__ q, float* __restrict__ k,
                      float* __restrict__ v) {
  int idx = blockIdx.x * blockDim.x + threadIdx.x;
  if (idx >= B_ * N2_ * EMB_) return;
  int d  = idx & 7;
  int bn = idx >> 3;
  const float* p = x2 + (size_t)bn * 3;
  float a0 = p[0], a1 = p[1], a2 = p[2];
  q[idx] = a0 * Wq[8 + d] + a1 * Wq[16 + d] + a2 * Wq[24 + d] + bq[d];
  k[idx] = a0 * Wk[8 + d] + a1 * Wk[16 + d] + a2 * Wk[24 + d] + bk[d];
  v[idx] = a0 * Wv[8 + d] + a1 * Wv[16 + d] + a2 * Wv[24 + d] + bv[d];
}

// ---------------------------------------------------------------------------
// 2) softmax attention over N2 keys; one wave32 per (b, n) query row
// ---------------------------------------------------------------------------
__global__ __launch_bounds__(128) void k_attn(const float* __restrict__ q,
                                              const float* __restrict__ k,
                                              const float* __restrict__ v,
                                              float* __restrict__ att) {
  __shared__ float sc[4][640];
  int wave = threadIdx.x >> 5, lane = threadIdx.x & 31;
  int bn = blockIdx.x * 4 + wave;
  if (bn >= B_ * N2_) return;                   // wave-uniform
  int b = bn / N2_;
  const float* qp = q + (size_t)bn * EMB_;
  float qr[8];
#pragma unroll
  for (int j = 0; j < 8; ++j) qr[j] = qp[j];
  const float* kb = k + (size_t)b * N2_ * EMB_;
  float lmax = -1e30f;
  for (int m = lane; m < N2_; m += 32) {
    const float* kp = kb + (size_t)m * EMB_;
    float s = 0.f;
#pragma unroll
    for (int j = 0; j < 8; ++j) s += qr[j] * kp[j];
    s *= 0.35355339059327373f;                  // 1/sqrt(8)
    sc[wave][m] = s;
    lmax = fmaxf(lmax, s);
  }
#pragma unroll
  for (int off = 16; off; off >>= 1) lmax = fmaxf(lmax, __shfl_xor(lmax, off, 32));
  const float* vb = v + (size_t)b * N2_ * EMB_;
  float lsum = 0.f;
  float av[8] = {0.f, 0.f, 0.f, 0.f, 0.f, 0.f, 0.f, 0.f};
  for (int m = lane; m < N2_; m += 32) {
    float p = __expf(sc[wave][m] - lmax);
    lsum += p;
    const float* vp = vb + (size_t)m * EMB_;
#pragma unroll
    for (int j = 0; j < 8; ++j) av[j] += p * vp[j];
  }
#pragma unroll
  for (int off = 16; off; off >>= 1) {
    lsum += __shfl_xor(lsum, off, 32);
#pragma unroll
    for (int j = 0; j < 8; ++j) av[j] += __shfl_xor(av[j], off, 32);
  }
  if (lane == 0) {
    float inv = 1.f / lsum;
    float* op = att + (size_t)bn * EMB_;
#pragma unroll
    for (int j = 0; j < 8; ++j) op[j] = av[j] * inv;
  }
}

// ---------------------------------------------------------------------------
// 3) assemble h0 [B, N3, KPAD] in bf16: concat(x, x_hidden, att), unpool
//    midpoints for n >= N2, zero K-pad tail.
// ---------------------------------------------------------------------------
__device__ __forceinline__ float feat_at(const float* x, const float* xh,
                                         const float* att, int b, int n, int c) {
  if (c < FEAT_)        return x  [((size_t)b * N2_ + n) * FEAT_ + c];
  if (c < FEAT_ + HID_) return xh [((size_t)b * N2_ + n) * HID_  + (c - FEAT_)];
  return                       att[((size_t)b * N2_ + n) * EMB_  + (c - FEAT_ - HID_)];
}

__global__ void k_build_h0(const float* __restrict__ x, const float* __restrict__ xh,
                           const float* __restrict__ att, const int* __restrict__ unp,
                           __bf16* __restrict__ h0) {
  int bm = blockIdx.x;
  int b = bm / N3_, n = bm - b * N3_;
  bool up = n >= N2_;
  int p0 = 0, p1 = 0;
  if (up) { p0 = unp[(n - N2_) * 2]; p1 = unp[(n - N2_) * 2 + 1]; }
  __bf16* dst = h0 + (size_t)bm * KPAD_;
  for (int c = threadIdx.x; c < KPAD_; c += blockDim.x) {
    float val;
    if (c >= IN_DIM_)      val = 0.f;
    else if (!up)          val = feat_at(x, xh, att, b, n, c);
    else                   val = 0.5f * (feat_at(x, xh, att, b, p0, c) +
                                         feat_at(x, xh, att, b, p1, c));
    dst[c] = (__bf16)val;
  }
}

// ---------------------------------------------------------------------------
// 4) weight prep: [K, N] fp32 -> [N, Kpad] bf16 (transpose + zero K-pad)
//    blockIdx.y selects matrix in a packed stack (res weights).
// ---------------------------------------------------------------------------
__global__ void k_wt_cvt(const float* __restrict__ src, __bf16* __restrict__ dst,
                         int K, int N, int Kpad) {
  int mi = blockIdx.y;
  src += (size_t)mi * K * N;
  dst += (size_t)mi * N * Kpad;
  int total = N * Kpad;
  for (int idx = blockIdx.x * blockDim.x + threadIdx.x; idx < total;
       idx += gridDim.x * blockDim.x) {
    int n = idx / Kpad, kk = idx - n * Kpad;
    float v = (kk < K) ? src[(size_t)kk * N + n] : 0.f;
    dst[idx] = (__bf16)v;
  }
}

// ---------------------------------------------------------------------------
// 5) adj -> fixed-width CSR (<= MAXD nnz/row; reference has <= 6)
// ---------------------------------------------------------------------------
__global__ void k_adj_extract(const float* __restrict__ adj, int* __restrict__ cols,
                              float* __restrict__ vals, int* __restrict__ cnt) {
  int m = blockIdx.x * blockDim.x + threadIdx.x;
  if (m >= N3_) return;
  const float* row = adj + (size_t)m * N3_;
  int c = 0;
  for (int j = 0; j < N3_; ++j) {
    float a = row[j];
    if (a != 0.f && c < MAXD_) {
      cols[m * MAXD_ + c] = j;
      vals[m * MAXD_ + c] = a;
      ++c;
    }
  }
  cnt[m] = c;
}

// ---------------------------------------------------------------------------
// 6) batched WMMA GEMM:  C[b] (MxN fp32) = A[b] (MxK bf16) @ Wt^T
//    Wt is [N][Kpad] bf16 (pre-transposed). Block 128x64, wave tile 32x32.
//    A and B fragments both contiguous per-half K run (k = 16*half + s).
// ---------------------------------------------------------------------------
__device__ __forceinline__ v16bf load_afrag(const __bf16* p) {
  // elements 0..15 = K {base..base+15} (contiguous, matches B fragment)
  v8bf lo = *(const v8bf*)p;
  v8bf hi = *(const v8bf*)(p + 8);
  v16bf f;
#pragma unroll
  for (int i = 0; i < 8; ++i) { f[i] = lo[i]; f[i + 8] = hi[i]; }
  return f;
}

__global__ __launch_bounds__(256) void k_gemm(const __bf16* __restrict__ A, int lda,
                                              long strideA,
                                              const __bf16* __restrict__ Wt, int ldw,
                                              float* __restrict__ C, int ldc,
                                              long strideC, int M, int Kpad) {
  int bz = blockIdx.z;
  int m0 = blockIdx.x * 128;
  int n0 = blockIdx.y * 64;
  int wave = threadIdx.x >> 5, lane = threadIdx.x & 31;
  int wm = wave & 3, wn = wave >> 2;
  int rowBase = m0 + wm * 32;
  int colBase = n0 + wn * 32;
  int lr = lane & 15, lh = lane >> 4;

  int r0 = rowBase + lr;      if (r0 > M - 1) r0 = M - 1;
  int r1 = rowBase + 16 + lr; if (r1 > M - 1) r1 = M - 1;
  const __bf16* a0p = A + (size_t)bz * strideA + (size_t)r0 * lda + lh * 16;
  const __bf16* a1p = A + (size_t)bz * strideA + (size_t)r1 * lda + lh * 16;
  const __bf16* b0p = Wt + (size_t)(colBase + lr) * ldw + lh * 16;
  const __bf16* b1p = Wt + (size_t)(colBase + 16 + lr) * ldw + lh * 16;

  v8f acc00 = {}, acc01 = {}, acc10 = {}, acc11 = {};
  for (int kk = 0; kk < Kpad; kk += 32) {
    v16bf a0 = load_afrag(a0p + kk);
    v16bf a1 = load_afrag(a1p + kk);
    v16bf w0 = *(const v16bf*)(b0p + kk);
    v16bf w1 = *(const v16bf*)(b1p + kk);
    acc00 = __builtin_amdgcn_wmma_f32_16x16x32_bf16(false, a0, false, w0, (short)0,
                                                    acc00, false, false);
    acc01 = __builtin_amdgcn_wmma_f32_16x16x32_bf16(false, a0, false, w1, (short)0,
                                                    acc01, false, false);
    acc10 = __builtin_amdgcn_wmma_f32_16x16x32_bf16(false, a1, false, w0, (short)0,
                                                    acc10, false, false);
    acc11 = __builtin_amdgcn_wmma_f32_16x16x32_bf16(false, a1, false, w1, (short)0,
                                                    acc11, false, false);
  }

  float* Cb = C + (size_t)bz * strideC;
  int c0 = colBase + lr, c1 = colBase + 16 + lr;
#pragma unroll
  for (int r = 0; r < 8; ++r) {
    int row = rowBase + lh * 8 + r;
    if (row < M) {
      Cb[(size_t)row * ldc + c0] = acc00[r];
      Cb[(size_t)row * ldc + c1] = acc01[r];
    }
    int row2 = row + 16;
    if (row2 < M) {
      Cb[(size_t)row2 * ldc + c0] = acc10[r];
      Cb[(size_t)row2 * ldc + c1] = acc11[r];
    }
  }
}

// ---------------------------------------------------------------------------
// 7) gconv epilogue: z = gather_adj(t0) + t1 + bias ; ReLU ; optional
//    residual z = 0.5*(h_old + z); write fp32 and/or bf16 copies.
// ---------------------------------------------------------------------------
__global__ __launch_bounds__(192) void k_combine(
    const float* __restrict__ t0, const float* __restrict__ t1,
    const float* __restrict__ bias, const int* __restrict__ cols,
    const float* __restrict__ vals, const int* __restrict__ cnt,
    const float* __restrict__ h_old, int do_relu, float* __restrict__ out_f,
    __bf16* __restrict__ out_b) {
  int bm = blockIdx.x;
  int b = bm / N3_, m = bm - b * N3_;
  int f = threadIdx.x;
  size_t e = (size_t)bm * HID_ + f;
  float z = t1[e] + bias[f];
  int c = cnt[m];
  const float* tb = t0 + (size_t)b * N3_ * HID_;
  for (int j = 0; j < c; ++j)
    z += vals[m * MAXD_ + j] * tb[(size_t)cols[m * MAXD_ + j] * HID_ + f];
  if (do_relu) z = fmaxf(z, 0.f);
  if (h_old) z = 0.5f * (h_old[e] + z);
  if (out_f) out_f[e] = z;
  if (out_b) out_b[e] = (__bf16)z;
}

// ---------------------------------------------------------------------------
// 8) coordinate head: u = h @ Wg, uL = h @ loopWg  (one wave per row)
// ---------------------------------------------------------------------------
__global__ __launch_bounds__(128) void k_coord_mm(const float* __restrict__ h,
                                                  const float* __restrict__ Wg,
                                                  const float* __restrict__ LWg,
                                                  float* __restrict__ u,
                                                  float* __restrict__ uL) {
  int wave = threadIdx.x >> 5, lane = threadIdx.x & 31;
  int bm = blockIdx.x * 4 + wave;
  if (bm >= B_ * N3_) return;
  const float* hp = h + (size_t)bm * HID_;
  float u0 = 0, u1 = 0, u2 = 0, l0 = 0, l1 = 0, l2 = 0;
  for (int f = lane; f < HID_; f += 32) {
    float hv = hp[f];
    u0 += hv * Wg[f * 3 + 0];  u1 += hv * Wg[f * 3 + 1];  u2 += hv * Wg[f * 3 + 2];
    l0 += hv * LWg[f * 3 + 0]; l1 += hv * LWg[f * 3 + 1]; l2 += hv * LWg[f * 3 + 2];
  }
#pragma unroll
  for (int off = 16; off; off >>= 1) {
    u0 += __shfl_xor(u0, off, 32); u1 += __shfl_xor(u1, off, 32);
    u2 += __shfl_xor(u2, off, 32); l0 += __shfl_xor(l0, off, 32);
    l1 += __shfl_xor(l1, off, 32); l2 += __shfl_xor(l2, off, 32);
  }
  if (lane == 0) {
    u [bm * 3 + 0] = u0; u [bm * 3 + 1] = u1; u [bm * 3 + 2] = u2;
    uL[bm * 3 + 0] = l0; uL[bm * 3 + 1] = l1; uL[bm * 3 + 2] = l2;
  }
}

__global__ void k_coord_out(const float* __restrict__ u, const float* __restrict__ uL,
                            const float* __restrict__ bg, const int* __restrict__ cols,
                            const float* __restrict__ vals, const int* __restrict__ cnt,
                            float* __restrict__ out) {
  int idx = blockIdx.x * blockDim.x + threadIdx.x;
  if (idx >= B_ * N3_ * 3) return;
  int j = idx % 3, bm = idx / 3;
  int b = bm / N3_, m = bm - b * N3_;
  float z = uL[idx] + bg[j];
  int c = cnt[m];
  const float* ub = u + (size_t)b * N3_ * 3;
  for (int d = 0; d < c; ++d)
    z += vals[m * MAXD_ + d] * ub[cols[m * MAXD_ + d] * 3 + j];
  out[idx] = z;
}

// ---------------------------------------------------------------------------
// Host orchestration
// ---------------------------------------------------------------------------
extern "C" void kernel_launch(void* const* d_in, const int* in_sizes, int n_in,
                              void* d_out, int out_size, void* d_ws, size_t ws_size,
                              hipStream_t stream) {
  (void)in_sizes; (void)n_in; (void)out_size; (void)ws_size;

  const float* x     = (const float*)d_in[0];
  const float* x2    = (const float*)d_in[1];
  const float* xh    = (const float*)d_in[2];
  const float* Wq    = (const float*)d_in[3];
  const float* bq    = (const float*)d_in[4];
  const float* Wk    = (const float*)d_in[5];
  const float* bk    = (const float*)d_in[6];
  const float* Wv    = (const float*)d_in[7];
  const float* bv    = (const float*)d_in[8];
  const float* adj   = (const float*)d_in[9];
  const int*   unp   = (const int*)  d_in[10];
  const float* W_in  = (const float*)d_in[11];
  const float* LW_in = (const float*)d_in[12];
  const float* b_in  = (const float*)d_in[13];
  const float* resW  = (const float*)d_in[14];
  const float* resLW = (const float*)d_in[15];
  const float* resB  = (const float*)d_in[16];
  const float* W_out = (const float*)d_in[17];
  const float* LW_out= (const float*)d_in[18];
  const float* b_out = (const float*)d_in[19];
  const float* Wg    = (const float*)d_in[20];
  const float* LWg   = (const float*)d_in[21];
  const float* bg    = (const float*)d_in[22];

  // ---- workspace carve-up (~250 MB; everything written before read) ----
  char* w = (char*)d_ws;
  size_t off = 0;
  auto alloc = [&](size_t bytes) -> void* {
    void* p = w + off;
    off += (bytes + 255) & ~(size_t)255;
    return p;
  };
  const size_t BN2 = (size_t)B_ * N2_;
  const size_t BN3 = (size_t)B_ * N3_;

  float*  qb   = (float*) alloc(BN2 * EMB_ * 4);
  float*  kb   = (float*) alloc(BN2 * EMB_ * 4);
  float*  vb   = (float*) alloc(BN2 * EMB_ * 4);
  float*  attb = (float*) alloc(BN2 * EMB_ * 4);
  __bf16* h0   = (__bf16*)alloc(BN3 * KPAD_ * 2);          // ~93 MB (L2-resident)
  __bf16* wtIn = (__bf16*)alloc((size_t)HID_ * KPAD_ * 2);
  __bf16* wtLI = (__bf16*)alloc((size_t)HID_ * KPAD_ * 2);
  __bf16* wtR  = (__bf16*)alloc((size_t)12 * HID_ * HID_ * 2);
  __bf16* wtRL = (__bf16*)alloc((size_t)12 * HID_ * HID_ * 2);
  __bf16* wtO  = (__bf16*)alloc((size_t)HID_ * HID_ * 2);
  __bf16* wtLO = (__bf16*)alloc((size_t)HID_ * HID_ * 2);
  float*  t0   = (float*) alloc(BN3 * HID_ * 4);
  float*  t1   = (float*) alloc(BN3 * HID_ * 4);
  float*  hf   = (float*) alloc(BN3 * HID_ * 4);
  __bf16* hb   = (__bf16*)alloc(BN3 * HID_ * 2);
  __bf16* yb   = (__bf16*)alloc(BN3 * HID_ * 2);
  float*  x4   = (float*) alloc(BN3 * HID_ * 4);
  int*    aCol = (int*)   alloc((size_t)N3_ * MAXD_ * 4);
  float*  aVal = (float*) alloc((size_t)N3_ * MAXD_ * 4);
  int*    aCnt = (int*)   alloc((size_t)N3_ * 4);
  float*  uG   = (float*) alloc(BN3 * 3 * 4);
  float*  uLG  = (float*) alloc(BN3 * 3 * 4);

  // ---- one-time-per-call prep (deterministic; re-done every call) ----
  k_wt_cvt<<<dim3(888, 1),  256, 0, stream>>>(W_in,  wtIn, IN_DIM_, HID_, KPAD_);
  k_wt_cvt<<<dim3(888, 1),  256, 0, stream>>>(LW_in, wtLI, IN_DIM_, HID_, KPAD_);
  k_wt_cvt<<<dim3(144, 12), 256, 0, stream>>>(resW,  wtR,  HID_, HID_, HID_);
  k_wt_cvt<<<dim3(144, 12), 256, 0, stream>>>(resLW, wtRL, HID_, HID_, HID_);
  k_wt_cvt<<<dim3(144, 1),  256, 0, stream>>>(W_out,  wtO,  HID_, HID_, HID_);
  k_wt_cvt<<<dim3(144, 1),  256, 0, stream>>>(LW_out, wtLO, HID_, HID_, HID_);
  k_adj_extract<<<(N3_ + 255) / 256, 256, 0, stream>>>(adj, aCol, aVal, aCnt);

  // ---- attention + feature assembly ----
  k_qkv<<<(int)((BN2 * EMB_ + 255) / 256), 256, 0, stream>>>(
      x2, Wq, bq, Wk, bk, Wv, bv, qb, kb, vb);
  k_attn<<<(int)((BN2 + 3) / 4), 128, 0, stream>>>(qb, kb, vb, attb);
  k_build_h0<<<(int)BN3, 256, 0, stream>>>(x, xh, attb, unp, h0);

  const dim3 gEntry((N3_ + 127) / 128, 192 / 64, B_);
  const dim3 gRes  ((N3_ + 127) / 128, 192 / 64, B_);
  const long sA0 = (long)N3_ * KPAD_;
  const long sAh = (long)N3_ * HID_;
  const long sC  = (long)N3_ * HID_;

  // ---- entry GConv: h = relu(adj@(h0 Win) + h0 LWin + b) ----
  k_gemm<<<gEntry, 256, 0, stream>>>(h0, KPAD_, sA0, wtIn, KPAD_, t0, HID_, sC, N3_, KPAD_);
  k_gemm<<<gEntry, 256, 0, stream>>>(h0, KPAD_, sA0, wtLI, KPAD_, t1, HID_, sC, N3_, KPAD_);
  k_combine<<<(int)BN3, HID_, 0, stream>>>(t0, t1, b_in, aCol, aVal, aCnt,
                                           nullptr, 1, hf, hb);

  // ---- 6 residual blocks ----
  for (int i = 0; i < NBLK_; ++i) {
    const __bf16* w0  = wtR  + (size_t)(i * 2 + 0) * HID_ * HID_;
    const __bf16* w0l = wtRL + (size_t)(i * 2 + 0) * HID_ * HID_;
    const __bf16* w1  = wtR  + (size_t)(i * 2 + 1) * HID_ * HID_;
    const __bf16* w1l = wtRL + (size_t)(i * 2 + 1) * HID_ * HID_;
    const float* b0 = resB + (size_t)(i * 2 + 0) * HID_;
    const float* b1 = resB + (size_t)(i * 2 + 1) * HID_;

    k_gemm<<<gRes, 256, 0, stream>>>(hb, HID_, sAh, w0,  HID_, t0, HID_, sC, N3_, HID_);
    k_gemm<<<gRes, 256, 0, stream>>>(hb, HID_, sAh, w0l, HID_, t1, HID_, sC, N3_, HID_);
    k_combine<<<(int)BN3, HID_, 0, stream>>>(t0, t1, b0, aCol, aVal, aCnt,
                                             nullptr, 1, nullptr, yb);
    k_gemm<<<gRes, 256, 0, stream>>>(yb, HID_, sAh, w1,  HID_, t0, HID_, sC, N3_, HID_);
    k_gemm<<<gRes, 256, 0, stream>>>(yb, HID_, sAh, w1l, HID_, t1, HID_, sC, N3_, HID_);
    k_combine<<<(int)BN3, HID_, 0, stream>>>(t0, t1, b1, aCol, aVal, aCnt,
                                             hf, 1, hf, hb);      // 0.5*(h + y), in place
  }

  // ---- exit GConv (ReLU) ----
  k_gemm<<<gRes, 256, 0, stream>>>(hb, HID_, sAh, wtO,  HID_, t0, HID_, sC, N3_, HID_);
  k_gemm<<<gRes, 256, 0, stream>>>(hb, HID_, sAh, wtLO, HID_, t1, HID_, sC, N3_, HID_);
  k_combine<<<(int)BN3, HID_, 0, stream>>>(t0, t1, b_out, aCol, aVal, aCnt,
                                           nullptr, 1, x4, nullptr);

  // ---- coordinate head (192 -> 3) in fp32 ----
  k_coord_mm<<<(int)((BN3 + 3) / 4), 128, 0, stream>>>(x4, Wg, LWg, uG, uLG);
  k_coord_out<<<(int)((BN3 * 3 + 255) / 256), 256, 0, stream>>>(
      uG, uLG, bg, aCol, aVal, aCnt, (float*)d_out);
}
